// HierarchicalPointEncoder_59768764891924
// MI455X (gfx1250) — compile-verified, matrix operand order checked by reading, not run on hardware
//
#include <hip/hip_runtime.h>
#include <hip/hip_bf16.h>
#include <math.h>

typedef __attribute__((ext_vector_type(16))) _Float16 v16h;
typedef __attribute__((ext_vector_type(8)))  float    v8f;

#define BB      8
#define NPTS    4096
#define NG      128
#define KNN     16
#define DD      384
#define NHEAD   6
#define HDIM    64
#define LDEPTH  4
#define EPSV    1e-5f
#define KSTEP   32

// ---- gfx1250 async global->LDS staging (guarded: degrades to sync path) ----
#if defined(__HIP_DEVICE_COMPILE__) && defined(__gfx1250__) && \
    __has_builtin(__builtin_amdgcn_global_load_async_to_lds_b128)
#define ASYNC_B128 1
#else
#define ASYNC_B128 0
#endif
#if defined(__HIP_DEVICE_COMPILE__) && defined(__gfx1250__) && \
    __has_builtin(__builtin_amdgcn_global_load_async_to_lds_b32)
#define ASYNC_B32 1
#else
#define ASYNC_B32 0
#endif
#if ASYNC_B128 || ASYNC_B32
typedef int v4i_vs __attribute__((vector_size(16)));
typedef __attribute__((address_space(1))) v4i_vs g_v4i;  // global int4
typedef __attribute__((address_space(3))) v4i_vs l_v4i;  // LDS int4
typedef __attribute__((address_space(1))) int    g_i32;  // global int
typedef __attribute__((address_space(3))) int    l_i32;  // LDS int
#endif

__device__ __forceinline__ float gelu_exact(float x) {
  return 0.5f * x * (1.0f + erff(x * 0.70710678118654752f));
}

// ---------------------------------------------------------------------------
// Generic WMMA GEMM:  C[M,N] = act( A[M,K] @ B (+bias) (+R) )
// BT=1: B stored [N,K] (linear's w) ; BT=0: B stored [K,N] (x @ W case)
// ACT: 0 = none, 1 = exact gelu
// block = 128 threads (4 waves); block tile 64(M) x 64(N).
// Panels staged in LDS as f32 (async copy when available); f16 cvt at fragment
// build, pairs merge into ds_load_b64.
// ---------------------------------------------------------------------------
template <int BT, int ACT>
__global__ __launch_bounds__(128) void gemm_wmma(
    const float* __restrict__ A, const float* __restrict__ Bm,
    const float* __restrict__ bias, const float* __restrict__ R,
    float* __restrict__ C, int M, int N, int K) {
  __shared__ float As32[64][KSTEP];  // [row][k]    8KB
  __shared__ float Bs32[64][KSTEP];  // [col(n)][k] 8KB
  const int tid  = threadIdx.x;
  const int wave = tid >> 5;
  const int lane = tid & 31;
  const int half = lane >> 4;
  const int lr   = lane & 15;
  const int mBlock = blockIdx.y * 64;
  const int nBlock = blockIdx.x * 64;
  const bool k4 = ((K & 3) == 0);
  const bool n4 = ((N & 3) == 0);

  v8f acc0 = {}, acc1 = {}, acc2 = {}, acc3 = {};

  for (int k0 = 0; k0 < K; k0 += KSTEP) {
    // ---- A panel ----
    for (int i = tid; i < 64 * (KSTEP / 4); i += 128) {
      int r = i >> 3;           // 8 chunks of 4 per row
      int kc = (i & 7) * 4;
      int gm = mBlock + r, gk = k0 + kc;
      if (k4 && gm < M && gk + 3 < K) {
        const float* src = A + (size_t)gm * K + gk;
#if ASYNC_B128
        __builtin_amdgcn_global_load_async_to_lds_b128((g_v4i*)src, (l_v4i*)&As32[r][kc], 0, 0);
#else
        *(float4*)&As32[r][kc] = *(const float4*)src;
#endif
      } else {
#pragma unroll
        for (int j = 0; j < 4; ++j) {
          int kx = gk + j;
          As32[r][kc + j] = (gm < M && kx < K) ? A[(size_t)gm * K + kx] : 0.f;
        }
      }
    }
    // ---- B panel ----
    if (BT) {  // [N,K]: same pattern as A
      for (int i = tid; i < 64 * (KSTEP / 4); i += 128) {
        int r = i >> 3;
        int kc = (i & 7) * 4;
        int gn = nBlock + r, gk = k0 + kc;
        if (k4 && gn < N && gk + 3 < K) {
          const float* src = Bm + (size_t)gn * K + gk;
#if ASYNC_B128
          __builtin_amdgcn_global_load_async_to_lds_b128((g_v4i*)src, (l_v4i*)&Bs32[r][kc], 0, 0);
#else
          *(float4*)&Bs32[r][kc] = *(const float4*)src;
#endif
        } else {
#pragma unroll
          for (int j = 0; j < 4; ++j) {
            int kx = gk + j;
            Bs32[r][kc + j] = (gn < N && kx < K) ? Bm[(size_t)gn * K + kx] : 0.f;
          }
        }
      }
    } else {  // [K,N]: coalesced along N, transposed LDS destination
      for (int i = tid; i < KSTEP * 16; i += 128) {
        int kk = i >> 4;
        int nc = (i & 15) * 4;
        int gk = k0 + kk, gn = nBlock + nc;
        if (n4 && gk < K && gn + 3 < N) {
          const float* src = Bm + (size_t)gk * N + gn;
#if ASYNC_B32
#pragma unroll
          for (int j = 0; j < 4; ++j)
            __builtin_amdgcn_global_load_async_to_lds_b32((g_i32*)(src + j), (l_i32*)&Bs32[nc + j][kk], 0, 0);
#else
          const float4 v = *(const float4*)src;
          Bs32[nc][kk]     = v.x;
          Bs32[nc + 1][kk] = v.y;
          Bs32[nc + 2][kk] = v.z;
          Bs32[nc + 3][kk] = v.w;
#endif
        } else {
#pragma unroll
          for (int j = 0; j < 4; ++j)
            Bs32[nc + j][kk] = (gk < K && gn + j < N) ? Bm[(size_t)gk * N + gn + j] : 0.f;
        }
      }
    }
    // prefetch next K panel (global_prefetch_b8 path on gfx1250)
    if (k0 + KSTEP < K) {
      int gm = mBlock + (tid & 63);
      if (gm < M) __builtin_prefetch(&A[(size_t)gm * K + k0 + KSTEP], 0, 0);
    }
#if ASYNC_B128 || ASYNC_B32
    asm volatile("s_wait_asynccnt 0x0" ::: "memory");
#endif
    __syncthreads();

    // ---- build fragments (ISA 7.12.2 16-bit 16x32 layout) and multiply ----
    v16h af;
#pragma unroll
    for (int g = 0; g < 8; ++g) {
      int kk = ((g & 3) * 2) + half * 8 + ((g >> 2) * 16);
      af[2 * g]     = (_Float16)As32[wave * 16 + lr][kk];
      af[2 * g + 1] = (_Float16)As32[wave * 16 + lr][kk + 1];
    }
#pragma unroll
    for (int nt = 0; nt < 4; ++nt) {
      v16h bf;
#pragma unroll
      for (int g = 0; g < 8; ++g) {
        int kk = ((g & 3) * 2) + half * 8 + ((g >> 2) * 16);
        bf[2 * g]     = (_Float16)Bs32[nt * 16 + lr][kk];
        bf[2 * g + 1] = (_Float16)Bs32[nt * 16 + lr][kk + 1];
      }
      if (nt == 0) acc0 = __builtin_amdgcn_wmma_f32_16x16x32_f16(false, af, false, bf, (short)0, acc0, false, false);
      if (nt == 1) acc1 = __builtin_amdgcn_wmma_f32_16x16x32_f16(false, af, false, bf, (short)0, acc1, false, false);
      if (nt == 2) acc2 = __builtin_amdgcn_wmma_f32_16x16x32_f16(false, af, false, bf, (short)0, acc2, false, false);
      if (nt == 3) acc3 = __builtin_amdgcn_wmma_f32_16x16x32_f16(false, af, false, bf, (short)0, acc3, false, false);
    }
    __syncthreads();
  }

  // ---- write back: f32 C/D layout: N = lane&15, M = e + 8*half ----
#pragma unroll
  for (int nt = 0; nt < 4; ++nt) {
    const v8f& a = (nt == 0) ? acc0 : (nt == 1) ? acc1 : (nt == 2) ? acc2 : acc3;
    int gn = nBlock + nt * 16 + lr;
    if (gn >= N) continue;
    float bv = bias ? bias[gn] : 0.f;
#pragma unroll
    for (int e = 0; e < 8; ++e) {
      int gm = mBlock + wave * 16 + e + half * 8;
      if (gm >= M) continue;
      float x = a[e] + bv;
      if (R) x += R[(size_t)gm * N + gn];
      if (ACT == 1) x = gelu_exact(x);
      C[(size_t)gm * N + gn] = x;
    }
  }
}

// ---------------------------------------------------------------------------
// WMMA attention: one block per (batch, head); 128 threads = 4 waves.
// S = (Q*scale) K^T via wmma, softmax in LDS (thread=row), O = P V via wmma.
// Static LDS = 16KB(Q) + 16KB(K) + 32KB(S) = 64KB; V^T reuses the Q buffer.
// ---------------------------------------------------------------------------
__global__ __launch_bounds__(128) void k_attention_wmma(const float* __restrict__ qkv,
                                                        float* __restrict__ o) {
  __shared__ _Float16 Qs[NG][HDIM];   // 16KB (later aliased by V^T [64][128])
  __shared__ _Float16 Ks[NG][HDIM];   // 16KB
  __shared__ _Float16 Sh[NG][NG];     // 32KB scores/probs
  const int b = blockIdx.x / NHEAD, h = blockIdx.x % NHEAD;
  const int tid = threadIdx.x, wave = tid >> 5, lane = tid & 31;
  const int half = lane >> 4, lr = lane & 15;
  const float* base = qkv + (size_t)b * NG * (3 * DD) + h * HDIM;

  for (int i = tid; i < NG * HDIM; i += 128) {
    int r = i >> 6, c = i & 63;
    Qs[r][c] = (_Float16)(base[(size_t)r * (3 * DD) + c] * 0.125f);
    Ks[r][c] = (_Float16)(base[(size_t)r * (3 * DD) + DD + c]);
  }
  __syncthreads();

  for (int kt = 0; kt < 8; ++kt) {
#pragma unroll
    for (int mt = 0; mt < 2; ++mt) {
      int q0 = wave * 32 + mt * 16;
      v8f acc = {};
#pragma unroll
      for (int ks = 0; ks < HDIM; ks += KSTEP) {
        v16h af, bf;
#pragma unroll
        for (int g = 0; g < 8; ++g) {
          int kk = ((g & 3) * 2) + half * 8 + ((g >> 2) * 16) + ks;
          af[2 * g]     = Qs[q0 + lr][kk];
          af[2 * g + 1] = Qs[q0 + lr][kk + 1];
          bf[2 * g]     = Ks[kt * 16 + lr][kk];
          bf[2 * g + 1] = Ks[kt * 16 + lr][kk + 1];
        }
        acc = __builtin_amdgcn_wmma_f32_16x16x32_f16(false, af, false, bf, (short)0, acc, false, false);
      }
#pragma unroll
      for (int e = 0; e < 8; ++e)
        Sh[q0 + e + 8 * half][kt * 16 + lr] = (_Float16)acc[e];
    }
  }
  __syncthreads();

  {
    float mx = -3.0e38f;
    for (int j = 0; j < NG; ++j) mx = fmaxf(mx, (float)Sh[tid][j]);
    float sum = 0.f;
    for (int j = 0; j < NG; ++j) sum += expf((float)Sh[tid][j] - mx);
    float inv = 1.f / sum;
    for (int j = 0; j < NG; ++j)
      Sh[tid][j] = (_Float16)(expf((float)Sh[tid][j] - mx) * inv);
  }
  __syncthreads();

  _Float16(*Vt)[NG] = reinterpret_cast<_Float16(*)[NG]>(&Qs[0][0]);
  for (int i = tid; i < HDIM * NG; i += 128) {
    int c = i >> 7, j = i & 127;
    Vt[c][j] = (_Float16)base[(size_t)j * (3 * DD) + 2 * DD + c];
  }
  __syncthreads();

#pragma unroll
  for (int mt = 0; mt < 2; ++mt) {
    int q0 = wave * 32 + mt * 16;
#pragma unroll
    for (int nt = 0; nt < 4; ++nt) {
      v8f acc = {};
      for (int ks = 0; ks < NG; ks += KSTEP) {
        v16h af, bf;
#pragma unroll
        for (int g = 0; g < 8; ++g) {
          int kk = ((g & 3) * 2) + half * 8 + ((g >> 2) * 16) + ks;
          af[2 * g]     = Sh[q0 + lr][kk];
          af[2 * g + 1] = Sh[q0 + lr][kk + 1];
          bf[2 * g]     = Vt[nt * 16 + lr][kk];
          bf[2 * g + 1] = Vt[nt * 16 + lr][kk + 1];
        }
        acc = __builtin_amdgcn_wmma_f32_16x16x32_f16(false, af, false, bf, (short)0, acc, false, false);
      }
#pragma unroll
      for (int e = 0; e < 8; ++e) {
        int row = b * NG + q0 + e + 8 * half;
        o[(size_t)row * DD + h * HDIM + nt * 16 + lr] = acc[e];
      }
    }
  }
}

// ---------------------------------------------------------------------------
// Stage kernels (VALU / reductions)
// ---------------------------------------------------------------------------

__global__ void k_feat0(const float* xyz, const float* w, const float* bias, float* f0) {
  int t = blockIdx.x * blockDim.x + threadIdx.x;
  if (t >= BB * NPTS) return;
  int b = t / NPTS, n = t % NPTS;
  const float* p = xyz + ((size_t)b * NPTS + n) * 7;
  for (int o = 0; o < 8; ++o)
    f0[((size_t)b * 8 + o) * NPTS + n] = w[o * 3] * p[0] + w[o * 3 + 1] * p[1] + w[o * 3 + 2] * p[2] + bias[o];
}

__global__ void k_knn(const float* qpts, int qs, const float* kpts, int ks,
                      int NQ, int NK, int* outIdx) {
  int t = blockIdx.x * blockDim.x + threadIdx.x;
  if (t >= BB * NQ) return;
  int b = t / NQ, qi = t % NQ;
  const float* q = qpts + ((size_t)b * NQ + qi) * qs;
  float qx = q[0], qy = q[1], qz = q[2];
  float bd[KNN]; int bi[KNN];
  for (int i = 0; i < KNN; ++i) { bd[i] = 3.0e38f; bi[i] = 0; }
  const float* kb = kpts + (size_t)b * NK * ks;
  for (int j = 0; j < NK; ++j) {
    const float* kp = kb + (size_t)j * ks;
    float dx = qx - kp[0], dy = qy - kp[1], dz = qz - kp[2];
    float d = dx * dx + dy * dy + dz * dz;
    if (d < bd[KNN - 1]) {
      int p = KNN - 1;
      while (p > 0 && bd[p - 1] > d) { bd[p] = bd[p - 1]; bi[p] = bi[p - 1]; --p; }
      bd[p] = d; bi[p] = j;
    }
  }
  int* o = outIdx + ((size_t)b * NQ + qi) * KNN;
  for (int i = 0; i < KNN; ++i) o[i] = bi[i];
}

__global__ void k_ec_stats(const float* fq, const float* fk, const int* idx, const float* w,
                           int IC, int OC, int G, int NQ, int NK, float* stats) {
  int b = blockIdx.x / G, g = blockIdx.x % G;
  int OCg = OC / G;
  __shared__ float ss[256], sq[256];
  float s = 0.f, s2 = 0.f;
  int total = OCg * NQ;
  for (int e = threadIdx.x; e < total; e += blockDim.x) {
    int oc = g * OCg + e / NQ;
    int n = e % NQ;
    const float* wr = w + (size_t)oc * 2 * IC;
    float tq = 0.f;
    for (int ic = 0; ic < IC; ++ic)
      tq += (wr[IC + ic] - wr[ic]) * fq[((size_t)b * IC + ic) * NQ + n];
    const int* nb = idx + ((size_t)b * NQ + n) * KNN;
    for (int k = 0; k < KNN; ++k) {
      int j = nb[k];
      float v = tq;
      for (int ic = 0; ic < IC; ++ic)
        v += wr[ic] * fk[((size_t)b * IC + ic) * NK + j];
      s += v; s2 += v * v;
    }
  }
  ss[threadIdx.x] = s; sq[threadIdx.x] = s2;
  __syncthreads();
  for (int st = blockDim.x / 2; st > 0; st >>= 1) {
    if (threadIdx.x < st) { ss[threadIdx.x] += ss[threadIdx.x + st]; sq[threadIdx.x] += sq[threadIdx.x + st]; }
    __syncthreads();
  }
  if (threadIdx.x == 0) {
    float cnt = (float)OCg * (float)NQ * (float)KNN;
    float mu = ss[0] / cnt;
    float var = sq[0] / cnt - mu * mu;
    stats[(b * G + g) * 2] = mu;
    stats[(b * G + g) * 2 + 1] = rsqrtf(var + EPSV);
  }
}

__global__ void k_ec_apply(const float* fq, const float* fk, const int* idx, const float* w,
                           const float* gamma, const float* beta, const float* stats,
                           int IC, int OC, int G, int NQ, int NK, float* out) {
  int t = blockIdx.x * blockDim.x + threadIdx.x;
  int per = OC * NQ;
  if (t >= BB * per) return;
  int b = t / per, oc = (t % per) / NQ, n = t % NQ;
  int OCg = OC / G, g = oc / OCg;
  float mu = stats[(b * G + g) * 2], rstd = stats[(b * G + g) * 2 + 1];
  const float* wr = w + (size_t)oc * 2 * IC;
  float tq = 0.f;
  for (int ic = 0; ic < IC; ++ic)
    tq += (wr[IC + ic] - wr[ic]) * fq[((size_t)b * IC + ic) * NQ + n];
  const int* nb = idx + ((size_t)b * NQ + n) * KNN;
  float m = -3.0e38f;
  for (int k = 0; k < KNN; ++k) {
    int j = nb[k];
    float v = tq;
    for (int ic = 0; ic < IC; ++ic)
      v += wr[ic] * fk[((size_t)b * IC + ic) * NK + j];
    v = (v - mu) * rstd * gamma[oc] + beta[oc];
    v = (v > 0.f) ? v : 0.2f * v;
    m = fmaxf(m, v);
  }
  out[((size_t)b * OC + oc) * NQ + n] = m;
}

__global__ void k_fps(const float* pts, int stride, int N, int M, float* dist, int* outIdx) {
  int b = blockIdx.x;
  const float* P = pts + (size_t)b * N * stride;
  float* db = dist + (size_t)b * N;
  __shared__ float sv[256];
  __shared__ int   si[256];
  for (int n = threadIdx.x; n < N; n += blockDim.x) db[n] = 1.0e10f;
  __syncthreads();
  int far = 0;
  for (int it = 0; it < M; ++it) {
    if (threadIdx.x == 0) outIdx[(size_t)b * M + it] = far;
    float cx = P[(size_t)far * stride], cy = P[(size_t)far * stride + 1], cz = P[(size_t)far * stride + 2];
    float bv = -1.f; int bidx = 0;
    for (int n = threadIdx.x; n < N; n += blockDim.x) {
      float dx = P[(size_t)n * stride] - cx;
      float dy = P[(size_t)n * stride + 1] - cy;
      float dz = P[(size_t)n * stride + 2] - cz;
      float d = dx * dx + dy * dy + dz * dz;
      float nd = fminf(db[n], d);
      db[n] = nd;
      if (nd > bv) { bv = nd; bidx = n; }
    }
    sv[threadIdx.x] = bv; si[threadIdx.x] = bidx;
    __syncthreads();
    for (int st = blockDim.x / 2; st > 0; st >>= 1) {
      if (threadIdx.x < st) {
        if (sv[threadIdx.x + st] > sv[threadIdx.x] ||
            (sv[threadIdx.x + st] == sv[threadIdx.x] && si[threadIdx.x + st] < si[threadIdx.x])) {
          sv[threadIdx.x] = sv[threadIdx.x + st]; si[threadIdx.x] = si[threadIdx.x + st];
        }
      }
      __syncthreads();
    }
    far = si[0];
    __syncthreads();
  }
}

__global__ void k_gather_ch(const float* in, const int* idx, int C, int Nin, int Nout, float* out) {
  int t = blockIdx.x * blockDim.x + threadIdx.x;
  if (t >= BB * C * Nout) return;
  int b = t / (C * Nout), c = (t % (C * Nout)) / Nout, i = t % Nout;
  out[((size_t)b * C + c) * Nout + i] = in[((size_t)b * C + c) * Nin + idx[(size_t)b * Nout + i]];
}

__global__ void k_gather_pts(const float* in, int stride, const int* idx, int Nin, int Nout, float* out) {
  int t = blockIdx.x * blockDim.x + threadIdx.x;
  if (t >= BB * Nout) return;
  int b = t / Nout, i = t % Nout;
  int j = idx[(size_t)b * Nout + i];
  const float* s = in + ((size_t)b * Nin + j) * stride;
  float* o = out + ((size_t)b * Nout + i) * 3;
  o[0] = s[0]; o[1] = s[1]; o[2] = s[2];
}

__global__ void k_cn2nc(const float* in, int C, int N, float* out) {
  int t = blockIdx.x * blockDim.x + threadIdx.x;
  if (t >= BB * C * N) return;
  int b = t / (C * N), c = (t % (C * N)) / N, n = t % N;
  out[((size_t)b * N + n) * C + c] = in[((size_t)b * C + c) * N + n];
}

__global__ void k_layernorm(const float* x, const float* g, const float* bb, int C, float* out) {
  int row = blockIdx.x;
  const float* xr = x + (size_t)row * C;
  __shared__ float ss[256], sq[256];
  __shared__ float s_mu, s_rs;
  float s = 0.f, s2 = 0.f;
  for (int c = threadIdx.x; c < C; c += blockDim.x) { float v = xr[c]; s += v; s2 += v * v; }
  ss[threadIdx.x] = s; sq[threadIdx.x] = s2;
  __syncthreads();
  for (int st = blockDim.x / 2; st > 0; st >>= 1) {
    if (threadIdx.x < st) { ss[threadIdx.x] += ss[threadIdx.x + st]; sq[threadIdx.x] += sq[threadIdx.x + st]; }
    __syncthreads();
  }
  if (threadIdx.x == 0) {
    float mu = ss[0] / C;
    float var = sq[0] / C - mu * mu;
    s_mu = mu; s_rs = rsqrtf(var + EPSV);
  }
  __syncthreads();
  float mu = s_mu, rs = s_rs;
  for (int c = threadIdx.x; c < C; c += blockDim.x)
    out[(size_t)row * C + c] = (xr[c] - mu) * rs * g[c] + bb[c];
}

__global__ void k_groupnorm_gelu(float* x, const float* gamma, const float* beta,
                                 int nTok, int C, int G, int doGelu) {
  int b = blockIdx.x / G, g = blockIdx.x % G;
  int Cg = C / G;
  __shared__ float ss[256], sq[256];
  __shared__ float s_mu, s_rs;
  float s = 0.f, s2 = 0.f;
  int total = nTok * Cg;
  for (int e = threadIdx.x; e < total; e += blockDim.x) {
    int n = e / Cg, c = g * Cg + e % Cg;
    float v = x[((size_t)b * nTok + n) * C + c];
    s += v; s2 += v * v;
  }
  ss[threadIdx.x] = s; sq[threadIdx.x] = s2;
  __syncthreads();
  for (int st = blockDim.x / 2; st > 0; st >>= 1) {
    if (threadIdx.x < st) { ss[threadIdx.x] += ss[threadIdx.x + st]; sq[threadIdx.x] += sq[threadIdx.x + st]; }
    __syncthreads();
  }
  if (threadIdx.x == 0) {
    float mu = ss[0] / total;
    float var = sq[0] / total - mu * mu;
    s_mu = mu; s_rs = rsqrtf(var + EPSV);
  }
  __syncthreads();
  float mu = s_mu, rs = s_rs;
  for (int e = threadIdx.x; e < total; e += blockDim.x) {
    int n = e / Cg, c = g * Cg + e % Cg;
    size_t off = ((size_t)b * nTok + n) * C + c;
    float v = (x[off] - mu) * rs * gamma[c] + beta[c];
    if (doGelu) v = gelu_exact(v);
    x[off] = v;
  }
}

__global__ void k_pool(const float* in, int nIn, int factor, float* out) {
  int nOut = nIn / factor;
  int t = blockIdx.x * blockDim.x + threadIdx.x;
  if (t >= BB * nOut * DD) return;
  int b = t / (nOut * DD), i = (t % (nOut * DD)) / DD, c = t % DD;
  float s = 0.f;
  for (int j = 0; j < factor; ++j) s += in[((size_t)b * nIn + i * factor + j) * DD + c];
  out[((size_t)b * nOut + i) * DD + c] = s / (float)factor;
}

__global__ void k_resize(const float* in, int nIn, int nOut, float* out) {
  int t = blockIdx.x * blockDim.x + threadIdx.x;
  if (t >= BB * nOut * DD) return;
  int b = t / (nOut * DD), i = (t % (nOut * DD)) / DD, c = t % DD;
  float pos = ((float)i + 0.5f) * ((float)nIn / (float)nOut) - 0.5f;
  pos = fminf(fmaxf(pos, 0.f), (float)nIn - 1.f);
  int i0 = (int)floorf(pos);
  int i1 = (i0 + 1 < nIn) ? i0 + 1 : nIn - 1;
  float w = pos - (float)i0;
  out[((size_t)b * nOut + i) * DD + c] =
      in[((size_t)b * nIn + i0) * DD + c] * (1.f - w) + in[((size_t)b * nIn + i1) * DD + c] * w;
}

__global__ void k_concat(const float* a, const float* b2, int M, int C, float* out) {
  int t = blockIdx.x * blockDim.x + threadIdx.x;
  if (t >= M * 2 * C) return;
  int r = t / (2 * C), c = t % (2 * C);
  out[(size_t)r * 2 * C + c] = (c < C) ? a[(size_t)r * C + c] : b2[(size_t)r * C + (c - C)];
}

__global__ void k_rownorm(const float* m, int rows, int C, float* out) {
  int r = blockIdx.x * blockDim.x + threadIdx.x;
  if (r >= rows) return;
  float s = 0.f;
  const float* p = m + (size_t)r * C;
  for (int c = 0; c < C; ++c) s += p[c] * p[c];
  out[r] = s;
}

__global__ void k_vq(const float* z, const float* dots, const float* cbn, const float* cb,
                     int Ncodes, int res, int rowOff, float* outBase, float* idxOut,
                     float* prevq, float* rowloss) {
  int row = blockIdx.x;
  int b = row / res, i = row % res;
  const float* zr = z + (size_t)row * DD;
  __shared__ float ss[256];
  __shared__ int   si[256];
  __shared__ float s_zn;
  __shared__ int   s_e;
  float s = 0.f;
  for (int c = threadIdx.x; c < DD; c += blockDim.x) { float v = zr[c]; s += v * v; }
  ss[threadIdx.x] = s;
  __syncthreads();
  for (int st = blockDim.x / 2; st > 0; st >>= 1) {
    if (threadIdx.x < st) ss[threadIdx.x] += ss[threadIdx.x + st];
    __syncthreads();
  }
  if (threadIdx.x == 0) s_zn = ss[0];
  __syncthreads();
  float zn = s_zn;
  const float* dr = dots + (size_t)row * Ncodes;
  float bv = 3.0e38f; int bidx = 0;
  for (int j = threadIdx.x; j < Ncodes; j += blockDim.x) {
    float d = zn - 2.f * dr[j] + cbn[j];
    if (d < bv) { bv = d; bidx = j; }
  }
  ss[threadIdx.x] = bv; si[threadIdx.x] = bidx;
  __syncthreads();
  for (int st = blockDim.x / 2; st > 0; st >>= 1) {
    if (threadIdx.x < st) {
      if (ss[threadIdx.x + st] < ss[threadIdx.x] ||
          (ss[threadIdx.x + st] == ss[threadIdx.x] && si[threadIdx.x + st] < si[threadIdx.x])) {
        ss[threadIdx.x] = ss[threadIdx.x + st]; si[threadIdx.x] = si[threadIdx.x + st];
      }
    }
    __syncthreads();
  }
  if (threadIdx.x == 0) s_e = si[0];
  __syncthreads();
  int e = s_e;
  const float* q = cb + (size_t)e * DD;
  float* dst = outBase + ((size_t)b * 224 + rowOff + i) * DD;
  float ls = 0.f;
  for (int c = threadIdx.x; c < DD; c += blockDim.x) {
    float qv = q[c];
    dst[c] = qv;
    prevq[(size_t)row * DD + c] = qv;
    float dlt = zr[c] - qv;
    ls += dlt * dlt;
  }
  ss[threadIdx.x] = ls;
  __syncthreads();
  for (int st = blockDim.x / 2; st > 0; st >>= 1) {
    if (threadIdx.x < st) ss[threadIdx.x] += ss[threadIdx.x + st];
    __syncthreads();
  }
  if (threadIdx.x == 0) { rowloss[row] = ss[0]; idxOut[row] = (float)e; }
}

__global__ void k_loss(const float* rl, float* out) {
  __shared__ float ss[256];
  float s = 0.f;
  for (int r = threadIdx.x; r < 1792; r += blockDim.x) {
    float w = (r < 1024) ? 1.f / (1024.f * 384.f)
            : (r < 1536) ? 1.f / (512.f * 384.f)
                         : 1.f / (256.f * 384.f);
    s += rl[r] * w;
  }
  ss[threadIdx.x] = s;
  __syncthreads();
  for (int st = blockDim.x / 2; st > 0; st >>= 1) {
    if (threadIdx.x < st) ss[threadIdx.x] += ss[threadIdx.x + st];
    __syncthreads();
  }
  if (threadIdx.x == 0) out[0] = 1.25f * ss[0];
}

// ---------------------------------------------------------------------------
// host side
// ---------------------------------------------------------------------------
static void gemm(hipStream_t s, const float* A, const float* Bm, const float* bias,
                 const float* R, float* C, int M, int N, int K, bool BT, int act) {
  dim3 g((N + 63) / 64, (M + 63) / 64), blk(128);
  if (BT) {
    if (act) gemm_wmma<1, 1><<<g, blk, 0, s>>>(A, Bm, bias, R, C, M, N, K);
    else     gemm_wmma<1, 0><<<g, blk, 0, s>>>(A, Bm, bias, R, C, M, N, K);
  } else {
    if (act) gemm_wmma<0, 1><<<g, blk, 0, s>>>(A, Bm, bias, R, C, M, N, K);
    else     gemm_wmma<0, 0><<<g, blk, 0, s>>>(A, Bm, bias, R, C, M, N, K);
  }
}

static inline dim3 g1(int n, int b) { return dim3((n + b - 1) / b); }

extern "C" void kernel_launch(void* const* d_in, const int* in_sizes, int n_in,
                              void* d_out, int out_size, void* d_ws, size_t ws_size,
                              hipStream_t stream) {
  const float* xyz   = (const float*)d_in[0];
  const float* it_w  = (const float*)d_in[1];
  const float* it_b  = (const float*)d_in[2];
  const float* g1_w  = (const float*)d_in[3];
  const float* g1_g  = (const float*)d_in[4];
  const float* g1_bt = (const float*)d_in[5];
  const float* g2_w  = (const float*)d_in[6];
  const float* g2_g  = (const float*)d_in[7];
  const float* g2_bt = (const float*)d_in[8];
  const float* g3_w  = (const float*)d_in[9];
  const float* g3_g  = (const float*)d_in[10];
  const float* g3_bt = (const float*)d_in[11];
  const float* g4_w  = (const float*)d_in[12];
  const float* g4_g  = (const float*)d_in[13];
  const float* g4_bt = (const float*)d_in[14];
  const float* pe_w1 = (const float*)d_in[15];
  const float* pe_b1 = (const float*)d_in[16];
  const float* pe_w2 = (const float*)d_in[17];
  const float* pe_b2 = (const float*)d_in[18];
  const float* ip_w1 = (const float*)d_in[19];
  const float* ip_b1 = (const float*)d_in[20];
  const float* ip_w2 = (const float*)d_in[21];
  const float* ip_b2 = (const float*)d_in[22];
  const float* ln1_g = (const float*)d_in[23];
  const float* ln1_b = (const float*)d_in[24];
  const float* qkv_w = (const float*)d_in[25];
  const float* qkv_b = (const float*)d_in[26];
  const float* o_w   = (const float*)d_in[27];
  const float* o_b   = (const float*)d_in[28];
  const float* ln2_g = (const float*)d_in[29];
  const float* ln2_b = (const float*)d_in[30];
  const float* m1_w  = (const float*)d_in[31];
  const float* m1_b  = (const float*)d_in[32];
  const float* m2_w  = (const float*)d_in[33];
  const float* m2_b  = (const float*)d_in[34];
  const float* pyr_w1  = (const float*)d_in[35];
  const float* pyr_b1  = (const float*)d_in[36];
  const float* pyr_g1  = (const float*)d_in[37];
  const float* pyr_bt1 = (const float*)d_in[38];
  const float* pyr_w2  = (const float*)d_in[39];
  const float* pyr_b2  = (const float*)d_in[40];
  const float* pyr_g2  = (const float*)d_in[41];
  const float* pyr_bt2 = (const float*)d_in[42];
  const float* fus_w = (const float*)d_in[43];
  const float* fus_b = (const float*)d_in[44];
  const float* cb0 = (const float*)d_in[45];
  const float* cb1 = (const float*)d_in[46];
  const float* cb2 = (const float*)d_in[47];

  float* dout = (float*)d_out;
  const size_t OUTS = (size_t)BB * 224 * DD;  // 688128

  char* wp = (char*)d_ws;
  auto take = [&](size_t bytes) -> void* {
    void* p = (void*)wp;
    wp += (bytes + 255) & ~(size_t)255;
    return p;
  };
  float* f0     = (float*)take((size_t)BB * 8 * NPTS * 4);
  float* f1     = (float*)take((size_t)BB * 32 * NPTS * 4);
  int*   knnIdx = (int*)  take((size_t)BB * NPTS * KNN * 4);
  float* fpsd   = (float*)take((size_t)BB * NPTS * 4);
  int*   fpsIdx = (int*)  take((size_t)BB * NG * 4);
  float* coorB  = (float*)take((size_t)BB * NG * 3 * 4);
  float* coorC  = (float*)take((size_t)BB * NG * 3 * 4);
  float* fq     = (float*)take((size_t)BB * 64 * NG * 4);
  float* f2     = (float*)take((size_t)BB * 64 * NG * 4);
  float* f3     = (float*)take((size_t)BB * 64 * NG * 4);
  float* f4     = (float*)take((size_t)BB * 128 * NG * 4);
  float* fT     = (float*)take((size_t)BB * NG * 128 * 4);
  float* ecst   = (float*)take(4096);
  float* tA     = (float*)take((size_t)BB * NG * 512 * 4);
  float* peb    = (float*)take((size_t)BB * NG * DD * 4);
  float* tbuf   = (float*)take((size_t)BB * NG * DD * 4);
  float* hn     = (float*)take((size_t)BB * NG * DD * 4);
  float* qkvb   = (float*)take((size_t)BB * NG * 3 * DD * 4);
  float* attno  = (float*)take((size_t)BB * NG * DD * 4);
  float* mlph   = (float*)take((size_t)BB * NG * 4 * DD * 4);
  float* lfb    = (float*)take((size_t)BB * NG * DD * 4);
  float* c1     = (float*)take((size_t)BB * NG * 2 * DD * 4);
  float* proc   = (float*)take((size_t)BB * NG * DD * 4);
  float* catb   = (float*)take((size_t)BB * NG * 2 * DD * 4);
  float* pu     = (float*)take((size_t)BB * NG * DD * 4);
  float* prevq  = (float*)take((size_t)BB * NG * DD * 4);
  float* cbn    = (float*)take(8192 * 4);
  float* rowls  = (float*)take(2048 * 4);
  float* dots   = (float*)take((size_t)BB * NG * 8192 * 4);
  (void)ws_size; (void)in_sizes; (void)n_in; (void)out_size;

  // ---- stage 1: input transform + edge convs + FPS ----
  k_feat0<<<g1(BB * NPTS, 256), 256, 0, stream>>>(xyz, it_w, it_b, f0);

  k_knn<<<g1(BB * NPTS, 128), 128, 0, stream>>>(xyz, 7, xyz, 7, NPTS, NPTS, knnIdx);
  k_ec_stats<<<BB * 4, 256, 0, stream>>>(f0, f0, knnIdx, g1_w, 8, 32, 4, NPTS, NPTS, ecst);
  k_ec_apply<<<g1(BB * 32 * NPTS, 256), 256, 0, stream>>>(f0, f0, knnIdx, g1_w, g1_g, g1_bt, ecst,
                                                          8, 32, 4, NPTS, NPTS, f1);

  k_fps<<<BB, 256, 0, stream>>>(xyz, 7, NPTS, NG, fpsd, fpsIdx);
  k_gather_pts<<<g1(BB * NG, 256), 256, 0, stream>>>(xyz, 7, fpsIdx, NPTS, NG, coorB);
  k_gather_ch<<<g1(BB * 32 * NG, 256), 256, 0, stream>>>(f1, fpsIdx, 32, NPTS, NG, fq);

  k_knn<<<g1(BB * NG, 128), 128, 0, stream>>>(coorB, 3, xyz, 7, NG, NPTS, knnIdx);
  k_ec_stats<<<BB * 4, 256, 0, stream>>>(fq, f1, knnIdx, g2_w, 32, 64, 4, NG, NPTS, ecst);
  k_ec_apply<<<g1(BB * 64 * NG, 256), 256, 0, stream>>>(fq, f1, knnIdx, g2_w, g2_g, g2_bt, ecst,
                                                        32, 64, 4, NG, NPTS, f2);

  k_knn<<<g1(BB * NG, 128), 128, 0, stream>>>(coorB, 3, coorB, 3, NG, NG, knnIdx);
  k_ec_stats<<<BB * 4, 256, 0, stream>>>(f2, f2, knnIdx, g3_w, 64, 64, 4, NG, NG, ecst);
  k_ec_apply<<<g1(BB * 64 * NG, 256), 256, 0, stream>>>(f2, f2, knnIdx, g3_w, g3_g, g3_bt, ecst,
                                                        64, 64, 4, NG, NG, f3);

  k_fps<<<BB, 256, 0, stream>>>(coorB, 3, NG, NG, fpsd, fpsIdx);
  k_gather_pts<<<g1(BB * NG, 256), 256, 0, stream>>>(coorB, 3, fpsIdx, NG, NG, coorC);
  k_gather_ch<<<g1(BB * 64 * NG, 256), 256, 0, stream>>>(f3, fpsIdx, 64, NG, NG, fq);

  k_knn<<<g1(BB * NG, 128), 128, 0, stream>>>(coorC, 3, coorB, 3, NG, NG, knnIdx);
  k_ec_stats<<<BB * 4, 256, 0, stream>>>(fq, f3, knnIdx, g4_w, 64, 128, 4, NG, NG, ecst);
  k_ec_apply<<<g1(BB * 128 * NG, 256), 256, 0, stream>>>(fq, f3, knnIdx, g4_w, g4_g, g4_bt, ecst,
                                                         64, 128, 4, NG, NG, f4);
  k_cn2nc<<<g1(BB * 128 * NG, 256), 256, 0, stream>>>(f4, 128, NG, fT);

  // ---- stage 2: embeddings (WMMA GEMMs) ----
  const int M = BB * NG;  // 1024 tokens
  gemm(stream, coorC, pe_w1, pe_b1, nullptr, tA, M, 128, 3, true, 1);
  gemm(stream, tA, pe_w2, pe_b2, nullptr, peb, M, DD, 128, true, 0);
  gemm(stream, fT, ip_w1, ip_b1, nullptr, tA, M, 512, 128, true, 1);
  gemm(stream, tA, ip_w2, ip_b2, peb, tbuf, M, DD, 512, true, 0);  // t = h + pe

  // ---- stage 3: transformer ----
  for (int l = 0; l < LDEPTH; ++l) {
    k_layernorm<<<M, 256, 0, stream>>>(tbuf, ln1_g + (size_t)l * DD, ln1_b + (size_t)l * DD, DD, hn);
    gemm(stream, hn, qkv_w + (size_t)l * DD * 3 * DD, qkv_b + (size_t)l * 3 * DD, nullptr,
         qkvb, M, 3 * DD, DD, false, 0);
    k_attention_wmma<<<BB * NHEAD, 128, 0, stream>>>(qkvb, attno);
    gemm(stream, attno, o_w + (size_t)l * DD * DD, o_b + (size_t)l * DD, tbuf, tbuf, M, DD, DD, false, 0);
    k_layernorm<<<M, 256, 0, stream>>>(tbuf, ln2_g + (size_t)l * DD, ln2_b + (size_t)l * DD, DD, hn);
    gemm(stream, hn, m1_w + (size_t)l * DD * 4 * DD, m1_b + (size_t)l * 4 * DD, nullptr,
         mlph, M, 4 * DD, DD, false, 1);
    gemm(stream, mlph, m2_w + (size_t)l * 4 * DD * DD, m2_b + (size_t)l * DD, tbuf, tbuf,
         M, DD, 4 * DD, false, 0);
  }

  // ---- stage 4: pyramid + VQ ----
  for (int level = 0; level < 3; ++level) {
    int res = NG >> level;  // 128, 64, 32
    int Ml = BB * res;
    const float* lf;
    if (level == 0) {
      lf = tbuf;
    } else {
      k_pool<<<g1(BB * res * DD, 256), 256, 0, stream>>>(tbuf, NG, NG / res, lfb);
      lf = lfb;
    }
    gemm(stream, lf, pyr_w1 + (size_t)level * 2 * DD * DD, pyr_b1 + (size_t)level * 2 * DD,
         nullptr, c1, Ml, 2 * DD, DD, true, 0);
    k_groupnorm_gelu<<<BB * 8, 256, 0, stream>>>(c1, pyr_g1 + (size_t)level * 2 * DD,
                                                 pyr_bt1 + (size_t)level * 2 * DD, res, 2 * DD, 8, 1);
    gemm(stream, c1, pyr_w2 + (size_t)level * DD * 2 * DD, pyr_b2 + (size_t)level * DD,
         nullptr, proc, Ml, DD, 2 * DD, true, 0);
    k_groupnorm_gelu<<<BB * 8, 256, 0, stream>>>(proc, pyr_g2 + (size_t)level * DD,
                                                 pyr_bt2 + (size_t)level * DD, res, DD, 8, 1);
    if (level > 0) {
      int nPrev = NG >> (level - 1);
      k_resize<<<g1(BB * res * DD, 256), 256, 0, stream>>>(prevq, nPrev, res, pu);
      k_concat<<<g1(Ml * 2 * DD, 256), 256, 0, stream>>>(proc, pu, Ml, DD, catb);
      gemm(stream, catb, fus_w + (size_t)(level - 1) * DD * 2 * DD, fus_b + (size_t)(level - 1) * DD,
           nullptr, proc, Ml, DD, 2 * DD, true, 1);
    }
    const float* cb = (level == 0) ? cb0 : (level == 1) ? cb1 : cb2;
    int NC = (level == 0) ? 8192 : (level == 1) ? 4096 : 2048;
    k_rownorm<<<g1(NC, 256), 256, 0, stream>>>(cb, NC, DD, cbn);
    gemm(stream, proc, cb, nullptr, nullptr, dots, Ml, NC, DD, true, 0);
    int rowOff = (level == 0) ? 0 : (level == 1) ? NG : NG + NG / 2;  // 0,128,192
    int lOff   = (level == 0) ? 0 : (level == 1) ? 1024 : 1536;
    float* idxOut = dout + OUTS + 1 + lOff;
    k_vq<<<Ml, 256, 0, stream>>>(proc, dots, cbn, cb, NC, res, rowOff, dout, idxOut, prevq,
                                 rowls + lOff);
  }

  k_loss<<<1, 256, 0, stream>>>(rowls, dout + OUTS);
}
